// AttentionMemoryEntry_78718160601441
// MI455X (gfx1250) — matrix operand order repaired and checked by reading, not compiled, not run on hardware
//
#include <hip/hip_runtime.h>

// ---------------------------------------------------------------------------
// Types
// ---------------------------------------------------------------------------
typedef __attribute__((ext_vector_type(16))) __bf16 v16bf;
typedef __attribute__((ext_vector_type(8)))  __bf16 v8bf;
typedef __attribute__((ext_vector_type(8)))  float  v8f;

__device__ __forceinline__ float bf2f(__bf16 x) {
    unsigned short s = __builtin_bit_cast(unsigned short, x);
    unsigned u = ((unsigned)s) << 16;
    return __builtin_bit_cast(float, u);
}
__device__ __forceinline__ __bf16 f2bf(float f) {
    unsigned u = __builtin_bit_cast(unsigned, f);
    unsigned r = u + 0x7FFFu + ((u >> 16) & 1u);   // round-to-nearest-even
    unsigned short s = (unsigned short)(r >> 16);
    return __builtin_bit_cast(__bf16, s);
}

// ---------------------------------------------------------------------------
// Problem constants
// ---------------------------------------------------------------------------
#define BB    8
#define LT    128
#define DD    512
#define HH    8
#define DH    64
#define DI    2048
#define NMEM  1024
#define LMEM  64
#define NQ    (BB*LT)        // 1024 queries
#define MKV   (NMEM*LMEM)    // 65536 rows of K/V

// ---------------------------------------------------------------------------
// LayerNorm over last dim (512). One wave32 per row.
// ---------------------------------------------------------------------------
__global__ __launch_bounds__(32)
void ln_kernel(const float* __restrict__ in, const float* __restrict__ g,
               const float* __restrict__ b, float* __restrict__ outF,
               __bf16* __restrict__ outB)
{
    const int row  = blockIdx.x;
    const int lane = threadIdx.x;
    const float* x = in + (size_t)row * DD;
    float v[16];
    float s = 0.f, s2 = 0.f;
#pragma unroll
    for (int i = 0; i < 16; ++i) {
        float t = x[lane + i * 32];
        v[i] = t; s += t; s2 += t * t;
    }
#pragma unroll
    for (int o = 16; o; o >>= 1) { s += __shfl_xor(s, o); s2 += __shfl_xor(s2, o); }
    float mu  = s * (1.f / DD);
    float var = s2 * (1.f / DD) - mu * mu;
    float inv = rsqrtf(var + 1e-5f);
#pragma unroll
    for (int i = 0; i < 16; ++i) {
        int d = lane + i * 32;
        float o = (v[i] - mu) * inv * g[d] + b[d];
        outF[(size_t)row * DD + d] = o;
        outB[(size_t)row * DD + d] = f2bf(o);
    }
}

// ---------------------------------------------------------------------------
// Row argmax over 1025 entries -> sample index - 1. One wave32 per row.
// ---------------------------------------------------------------------------
__global__ __launch_bounds__(32)
void argmax_kernel(const float* __restrict__ a, int* __restrict__ samples)
{
    const int row  = blockIdx.x;
    const int lane = threadIdx.x;
    const float* r = a + (size_t)row * (NMEM + 1);
    float bv = -3.4e38f; int bi = 0x7fffffff;
    for (int i = lane; i < NMEM + 1; i += 32) {
        float v = r[i];
        if (v > bv || (v == bv && i < bi)) { bv = v; bi = i; }
    }
#pragma unroll
    for (int o = 16; o; o >>= 1) {
        float ov = __shfl_xor(bv, o); int oi = __shfl_xor(bi, o);
        if (ov > bv || (ov == bv && oi < bi)) { bv = ov; bi = oi; }
    }
    if (lane == 0) samples[row] = bi - 1;
}

// ---------------------------------------------------------------------------
// Pack a row-major fp32 weight [K x N] into WMMA B-fragment order (bf16).
// Fragment (kt, nt) holds a 32x16 tile; lane L: n = L%16, kbase = (L/16)*16;
// element j (0..15): value = W[kt*32 + kbase + j][nt*16 + n]. 32 B/lane.
// ---------------------------------------------------------------------------
__global__ __launch_bounds__(256)
void pack_w_kernel(const float* __restrict__ W, __bf16* __restrict__ out,
                   int K, int N)
{
    int idx = blockIdx.x * 256 + threadIdx.x;
    if (idx >= K * N) return;
    int frag = idx >> 9;          // 512 elements per 32x16 fragment
    int r    = idx & 511;
    int lane = r >> 4;
    int j    = r & 15;
    int ntiles = N >> 4;
    int kt = frag / ntiles, nt = frag % ntiles;
    int n     = lane & 15;
    int kbase = (lane >> 4) * 16;
    out[idx] = f2bf(W[(size_t)(kt * 32 + kbase + j) * N + (nt * 16 + n)]);
}

// ---------------------------------------------------------------------------
// WMMA bf16 GEMM: C[M,N] = A[M,K] * B[K,N] (+bias)(+relu)(+fp32 residual)
// A: row-major, fp32 (converted inline) or bf16.  B: pre-packed fragments.
// Block = 128 threads (4 waves). Wave tile = (16*AF) M x 64 N.
// K double-buffered ping-pong (K % 64 == 0): loads for k+32 issued before
// the WMMAs of k, so s_wait_loadcnt becomes partial and latency is hidden.
// ---------------------------------------------------------------------------
template <bool A_BF16, bool RELU, int AF>
__global__ __launch_bounds__(128)
void gemm_wmma_kernel(const void* __restrict__ Araw,
                      const __bf16* __restrict__ Bp,
                      const float* __restrict__ bias,
                      const float* __restrict__ resid,
                      float* __restrict__ outF,
                      __bf16* __restrict__ outB,
                      int M, int N, int K)
{
    const int lane  = threadIdx.x & 31;
    const int wave  = threadIdx.x >> 5;
    const int mBase = (blockIdx.y * 4 + wave) * (16 * AF);
    const int nBase = blockIdx.x * 64;
    const int mrow  = lane & 15;
    const int khalf = (lane >> 4) * 16;          // A-fragment K sub-offset
    const int ntiles = N >> 4;
    const size_t bstep = (size_t)ntiles * 512;   // packed-B bytes(elems)/K-tile
    const __bf16* BpBase = Bp + (size_t)(nBase >> 4) * 512 + lane * 16;

    v8f acc[AF][4];
#pragma unroll
    for (int f = 0; f < AF; ++f)
#pragma unroll
        for (int t = 0; t < 4; ++t)
#pragma unroll
            for (int r = 0; r < 8; ++r) acc[f][t][r] = 0.f;

    auto loadA = [&](int kk, v16bf (&a)[AF]) {
#pragma unroll
        for (int f = 0; f < AF; ++f) {
            const int m = mBase + f * 16 + mrow;
            if (A_BF16) {
                const __bf16* base = (const __bf16*)Araw + (size_t)m * K + kk;
                if (kk + 128 < K) __builtin_prefetch(base + 128, 0, 3);
                v8bf lo = *(const v8bf*)(base + khalf);
                v8bf hi = *(const v8bf*)(base + 8 + khalf);
                a[f] = __builtin_shufflevector(lo, hi,
                        0,1,2,3,4,5,6,7,8,9,10,11,12,13,14,15);
            } else {
                const float* base = (const float*)Araw + (size_t)m * K + kk;
                if (kk + 128 < K) __builtin_prefetch(base + 128, 0, 3);
                float4 f0 = *(const float4*)(base + khalf);
                float4 f1 = *(const float4*)(base + khalf + 4);
                float4 f2 = *(const float4*)(base + 8 + khalf);
                float4 f3 = *(const float4*)(base + 8 + khalf + 4);
                v16bf v;
                v[0]  = f2bf(f0.x); v[1]  = f2bf(f0.y); v[2]  = f2bf(f0.z); v[3]  = f2bf(f0.w);
                v[4]  = f2bf(f1.x); v[5]  = f2bf(f1.y); v[6]  = f2bf(f1.z); v[7]  = f2bf(f1.w);
                v[8]  = f2bf(f2.x); v[9]  = f2bf(f2.y); v[10] = f2bf(f2.z); v[11] = f2bf(f2.w);
                v[12] = f2bf(f3.x); v[13] = f2bf(f3.y); v[14] = f2bf(f3.z); v[15] = f2bf(f3.w);
                a[f] = v;
            }
        }
    };
    auto loadB = [&](int kk, v16bf (&b)[4]) {
        const __bf16* bb = BpBase + (size_t)(kk >> 5) * bstep;
#pragma unroll
        for (int t = 0; t < 4; ++t) {
            v8bf lo = *(const v8bf*)(bb + (size_t)t * 512);
            v8bf hi = *(const v8bf*)(bb + (size_t)t * 512 + 8);
            b[t] = __builtin_shufflevector(lo, hi,
                    0,1,2,3,4,5,6,7,8,9,10,11,12,13,14,15);
        }
    };
    auto mm = [&](v16bf (&a)[AF], v16bf (&b)[4]) {
#pragma unroll
        for (int f = 0; f < AF; ++f)
#pragma unroll
            for (int t = 0; t < 4; ++t)
                acc[f][t] = __builtin_amdgcn_wmma_f32_16x16x32_bf16(
                    false, a[f], false, b[t], (short)0, acc[f][t], false, false);
    };

    v16bf a0[AF], a1[AF], b0[4], b1[4];
    loadA(0, a0); loadB(0, b0);
    for (int kk = 0; kk < K; kk += 64) {          // K % 64 == 0 for all our GEMMs
        loadA(kk + 32, a1); loadB(kk + 32, b1);   // issue next loads first
        mm(a0, b0);
        if (kk + 64 < K) { loadA(kk + 64, a0); loadB(kk + 64, b0); }
        mm(a1, b1);
    }

    // ---- Epilogue: C layout — lane n = lane%16, row = r + (lane>=16)*8 ----
    const int cn   = lane & 15;
    const int rOff = (lane >> 4) * 8;
#pragma unroll
    for (int f = 0; f < AF; ++f) {
#pragma unroll
        for (int t = 0; t < 4; ++t) {
            int col = nBase + t * 16 + cn;
            float bi = bias ? bias[col] : 0.f;
#pragma unroll
            for (int r = 0; r < 8; ++r) {
                int row = mBase + f * 16 + rOff + r;
                float v = acc[f][t][r] + bi;
                if (RELU) v = v > 0.f ? v : 0.f;
                size_t o = (size_t)row * N + col;
                if (resid) v += resid[o];
                if (outF)  outF[o] = v;
                if (outB)  outB[o] = f2bf(v);
            }
        }
    }
}

// ---------------------------------------------------------------------------
// Per-query masked attention over the selected memory (64 keys, 8 heads).
// One 128-thread block per query. Tiny FLOPs; VALU is fine.
// ---------------------------------------------------------------------------
__global__ __launch_bounds__(128)
void attn_kernel(const __bf16* __restrict__ qh, const __bf16* __restrict__ Kall,
                 const __bf16* __restrict__ Vall, const int* __restrict__ samples,
                 const int* __restrict__ maskmem, __bf16* __restrict__ ctx)
{
    __shared__ float q[DD];
    __shared__ float sc[HH * LMEM];
    const int n = blockIdx.x;
    const int t = threadIdx.x;
    int s  = samples[n];
    int sm = s < 0 ? 0 : s;      // clamp; invalid queries zeroed later

    for (int d = t; d < DD; d += 128) q[d] = bf2f(qh[(size_t)n * DD + d]);
    __syncthreads();

    if (t < LMEM) {
        const int k = t;
        const __bf16* Kr = Kall + ((size_t)sm * LMEM + k) * DD;
        const bool km = maskmem[sm * LMEM + k] != 0;
#pragma unroll
        for (int h = 0; h < HH; ++h) {
            float acc = 0.f;
#pragma unroll 8
            for (int d = 0; d < DH; ++d) acc += q[h * DH + d] * bf2f(Kr[h * DH + d]);
            sc[h * LMEM + k] = km ? acc * 0.125f : -1e9f;
        }
    }
    __syncthreads();
    if (t < HH) {
        const int h = t;
        float mx = -3.4e38f;
        for (int k = 0; k < LMEM; ++k) mx = fmaxf(mx, sc[h * LMEM + k]);
        float sum = 0.f;
        for (int k = 0; k < LMEM; ++k) {
            float e = __expf(sc[h * LMEM + k] - mx);
            sc[h * LMEM + k] = e; sum += e;
        }
        float inv = 1.f / sum;
        for (int k = 0; k < LMEM; ++k) sc[h * LMEM + k] *= inv;
    }
    __syncthreads();
    for (int d = t; d < DD; d += 128) {
        const int h = d >> 6;
        float acc = 0.f;
#pragma unroll 8
        for (int k = 0; k < LMEM; ++k)
            acc += sc[h * LMEM + k] * bf2f(Vall[((size_t)sm * LMEM + k) * DD + d]);
        ctx[(size_t)n * DD + d] = f2bf(acc);
    }
}

// ---------------------------------------------------------------------------
// y = x + (valid ? LN(st2) : 0)   — one wave32 per row
// ---------------------------------------------------------------------------
__global__ __launch_bounds__(32)
void ln_combine_kernel(const float* __restrict__ st2, const float* __restrict__ g,
                       const float* __restrict__ b, const float* __restrict__ x,
                       const int* __restrict__ samples,
                       float* __restrict__ yF, __bf16* __restrict__ yB)
{
    const int row  = blockIdx.x;
    const int lane = threadIdx.x;
    const bool valid = samples[row] >= 0;
    const float* in = st2 + (size_t)row * DD;
    float v[16];
    float s = 0.f, s2 = 0.f;
#pragma unroll
    for (int i = 0; i < 16; ++i) {
        float t = in[lane + i * 32];
        v[i] = t; s += t; s2 += t * t;
    }
#pragma unroll
    for (int o = 16; o; o >>= 1) { s += __shfl_xor(s, o); s2 += __shfl_xor(s2, o); }
    float mu  = s * (1.f / DD);
    float var = s2 * (1.f / DD) - mu * mu;
    float inv = rsqrtf(var + 1e-5f);
#pragma unroll
    for (int i = 0; i < 16; ++i) {
        int d = lane + i * 32;
        float o = (v[i] - mu) * inv * g[d] + b[d];
        float y = x[(size_t)row * DD + d] + (valid ? o : 0.f);
        yF[(size_t)row * DD + d] = y;
        yB[(size_t)row * DD + d] = f2bf(y);
    }
}

// ---------------------------------------------------------------------------
// Host-side launcher
// ---------------------------------------------------------------------------
extern "C" void kernel_launch(void* const* d_in, const int* in_sizes, int n_in,
                              void* d_out, int out_size, void* d_ws, size_t ws_size,
                              hipStream_t stream)
{
    const float* dec      = (const float*)d_in[0];
    const float* mem_attn = (const float*)d_in[1];
    const float* enc      = (const float*)d_in[2];
    const float* tgt      = (const float*)d_in[3];
    const int*   maskmem  = (const int*)  d_in[4];
    const float* g0  = (const float*)d_in[6];
    const float* be0 = (const float*)d_in[7];
    const float* g1  = (const float*)d_in[8];
    const float* be1 = (const float*)d_in[9];
    const float* Wq = (const float*)d_in[10]; const float* bq = (const float*)d_in[11];
    const float* Wk = (const float*)d_in[12]; const float* bk = (const float*)d_in[13];
    const float* Wv = (const float*)d_in[14]; const float* bv = (const float*)d_in[15];
    const float* Wo = (const float*)d_in[16]; const float* bo = (const float*)d_in[17];
    const float* f1W1 = (const float*)d_in[18]; const float* f1b1 = (const float*)d_in[19];
    const float* f1W2 = (const float*)d_in[20]; const float* f1b2 = (const float*)d_in[21];
    const float* f2W1 = (const float*)d_in[22]; const float* f2b1 = (const float*)d_in[23];
    const float* f2W2 = (const float*)d_in[24]; const float* f2b2 = (const float*)d_in[25];

    char* ws = (char*)d_ws;
    auto alloc = [&](size_t bytes) {
        char* p = ws;
        ws += (bytes + 255) & ~(size_t)255;
        return (void*)p;
    };
    float*  x_f32   = (float*) alloc((size_t)NQ * DD * 4);
    __bf16* x_bf16  = (__bf16*)alloc((size_t)NQ * DD * 2);
    int*    samples = (int*)   alloc((size_t)NQ * 4);
    __bf16* Wq_p   = (__bf16*)alloc((size_t)DD * DD * 2);
    __bf16* Wk_p   = (__bf16*)alloc((size_t)DD * DD * 2);
    __bf16* Wv_p   = (__bf16*)alloc((size_t)DD * DD * 2);
    __bf16* Wo_p   = (__bf16*)alloc((size_t)DD * DD * 2);
    __bf16* f1W1_p = (__bf16*)alloc((size_t)DD * DI * 2);
    __bf16* f1W2_p = (__bf16*)alloc((size_t)DI * DD * 2);
    __bf16* f2W1_p = (__bf16*)alloc((size_t)DD * DI * 2);
    __bf16* f2W2_p = (__bf16*)alloc((size_t)DI * DD * 2);
    __bf16* Kall   = (__bf16*)alloc((size_t)MKV * DD * 2);
    __bf16* Vall   = (__bf16*)alloc((size_t)MKV * DD * 2);
    __bf16* qh     = (__bf16*)alloc((size_t)NQ * DD * 2);
    __bf16* ctx    = (__bf16*)alloc((size_t)NQ * DD * 2);
    float*  st_f32 = (float*) alloc((size_t)NQ * DD * 4);
    __bf16* st_b16 = (__bf16*)alloc((size_t)NQ * DD * 2);
    __bf16* h1     = (__bf16*)alloc((size_t)NQ * DI * 2);
    float*  st2    = (float*) alloc((size_t)NQ * DD * 4);
    float*  y_f32  = (float*) alloc((size_t)NQ * DD * 4);
    __bf16* y_b16  = (__bf16*)alloc((size_t)NQ * DD * 2);
    __bf16* h2     = (__bf16*)alloc((size_t)NQ * DI * 2);

    // --- Stage 0: LN(dec_output), argmax samples, weight packing ---
    ln_kernel<<<NQ, 32, 0, stream>>>(dec, g0, be0, x_f32, x_bf16);
    argmax_kernel<<<NQ, 32, 0, stream>>>(mem_attn, samples);

    auto pack = [&](const float* W, __bf16* out, int K, int N) {
        int total = K * N;
        pack_w_kernel<<<(total + 255) / 256, 256, 0, stream>>>(W, out, K, N);
    };
    pack(Wq, Wq_p, DD, DD);   pack(Wk, Wk_p, DD, DD);
    pack(Wv, Wv_p, DD, DD);   pack(Wo, Wo_p, DD, DD);
    pack(f1W1, f1W1_p, DD, DI); pack(f1W2, f1W2_p, DI, DD);
    pack(f2W1, f2W1_p, DD, DI); pack(f2W2, f2W2_p, DI, DD);

    auto gemm = [&](const void* A, bool a_bf16, const __bf16* Bp, const float* bias,
                    const float* resid, float* outF, __bf16* outB,
                    int M, int N, int K, bool relu, int af) {
        dim3 block(128);
        if (!a_bf16 && af == 2) {
            dim3 grid(N / 64, M / 128);
            gemm_wmma_kernel<false, false, 2><<<grid, block, 0, stream>>>(A, Bp, bias, resid, outF, outB, M, N, K);
        } else if (a_bf16 && relu) {
            dim3 grid(N / 64, M / 64);
            gemm_wmma_kernel<true, true, 1><<<grid, block, 0, stream>>>(A, Bp, bias, resid, outF, outB, M, N, K);
        } else if (a_bf16) {
            dim3 grid(N / 64, M / 64);
            gemm_wmma_kernel<true, false, 1><<<grid, block, 0, stream>>>(A, Bp, bias, resid, outF, outB, M, N, K);
        } else {
            dim3 grid(N / 64, M / 64);
            gemm_wmma_kernel<false, false, 1><<<grid, block, 0, stream>>>(A, Bp, bias, resid, outF, outB, M, N, K);
        }
    };

    // --- Stage 1: dense K/V projection of ALL memories (the 69-GFLOP core).
    //     AF=2: 32 M-rows/wave -> 8 WMMAs per K-step per wave, double-buffered.
    gemm(enc, false, Wk_p, bk, nullptr, nullptr, Kall, MKV, DD, DD, false, 2);
    gemm(tgt, false, Wv_p, bv, nullptr, nullptr, Vall, MKV, DD, DD, false, 2);

    // --- Stage 2: Q projection ---
    gemm(x_bf16, true, Wq_p, bq, nullptr, nullptr, qh, NQ, DD, DD, false, 1);

    // --- Stage 3: per-query masked attention ---
    attn_kernel<<<NQ, 128, 0, stream>>>(qh, Kall, Vall, samples, maskmem, ctx);

    // --- Stage 4: st = q + ctx@Wo + bo ---
    gemm(ctx, true, Wo_p, bo, x_f32, st_f32, st_b16, NQ, DD, DD, false, 1);

    // --- Stage 5: FFN1 with residual ---
    gemm(st_b16, true, f1W1_p, f1b1, nullptr, nullptr, h1, NQ, DI, DD, true, 1);
    gemm(h1, true, f1W2_p, f1b2, st_f32, st2, nullptr, NQ, DD, DI, false, 1);

    // --- Stage 6: y = x + (valid ? LN(st2) : 0) ---
    ln_combine_kernel<<<NQ, 32, 0, stream>>>(st2, g1, be1, x_f32, samples, y_f32, y_b16);

    // --- Stage 7: FFN2 with residual -> output ---
    gemm(y_b16, true, f2W1_p, f2b1, nullptr, nullptr, h2, NQ, DI, DD, true, 1);
    gemm(h2, true, f2W2_p, f2b2, y_f32, (float*)d_out, nullptr, NQ, DD, DI, false, 1);
}
